// GraphConvNorm_77695958385178
// MI455X (gfx1250) — matrix operand order repaired and checked by reading, not compiled, not run on hardware
//
#include <hip/hip_runtime.h>
#include <math.h>

// ---------------------------------------------------------------------------
// GraphConvNorm on MI455X (gfx1250, wave32)
//
// Strategy (memory-bound op, 23.3 TB/s HBM):
//   - Segment sums kept in bf16 (179 MB instead of 358 MB f32); accumulated
//     with native global_atomic_pk_add_bf16 (inline asm, CDNA5 VMEM atomic).
//   - Edges with edge_type==6 skipped (slot overwritten by self features).
//   - GEMM [N,896]x[896,128] via v_wmma_f32_16x16x32_bf16 (f32 accumulate),
//     1/cnt scaling + self-row substitution fused into the A-fragment build.
//   - W pre-transposed+bf16-packed so each B fragment is ONE 32-byte load.
//   - BatchNorm: per-channel sum/sumsq via atomics, then normalize in place.
//
// Workspace layout (bytes, all 32B-aligned):
//   [0, 7N*128*2)              bf16 segment sums
//   [+, 7N*4)                  u32 segment counts
//   [+, 256*4)                 BN stats (sum, sumsq per channel)
//   [+, 128*896*2)             Wt: transposed bf16 weights  (~182.2 MB total)
// ---------------------------------------------------------------------------

typedef __attribute__((ext_vector_type(16))) __bf16 v16bf;
typedef __attribute__((ext_vector_type(8)))  __bf16 v8bf;
typedef __attribute__((ext_vector_type(8)))  float  v8f;

#define C_DIM 128
#define K_DIM 896   // 7 * 128

// ---- transpose + bf16-pack weights: Wt[n*896 + k] = bf16(W[k*128 + n]) ----
__global__ __launch_bounds__(256) void pack_w_kernel(const float* __restrict__ W,
                                                     unsigned short* __restrict__ Wt) {
    int i = blockIdx.x * 256 + threadIdx.x;
    if (i >= K_DIM * C_DIM) return;
    int k = i / C_DIM, n = i % C_DIM;
    union { __bf16 b; unsigned short u; } cv;
    cv.b = (__bf16)W[(size_t)k * C_DIM + n];
    Wt[(size_t)n * K_DIM + k] = cv.u;
}

// ---- edge scatter: one wave per edge; 4 channels/lane; pk_add_bf16 atomics ----
__global__ __launch_bounds__(256) void scatter_kernel(const float* __restrict__ x,
                                                      const int* __restrict__ row,
                                                      const int* __restrict__ col,
                                                      const int* __restrict__ et,
                                                      unsigned short* __restrict__ sums,
                                                      unsigned* __restrict__ cnt, int E) {
    int gw   = (blockIdx.x * 256 + threadIdx.x) >> 5;   // wave id == edge id
    int lane = threadIdx.x & 31;
    if (gw >= E) return;
    int t = et[gw];
    if (t == 6) return;                     // slot 6 is overwritten by x -> dead work
    int r = row[gw], c = col[gw];
    size_t seg = (size_t)r * 7 + (size_t)t;
    if (lane == 0) atomicAdd(&cnt[seg], 1u);
    const float4 v = *(const float4*)(x + (size_t)c * C_DIM + lane * 4);
    union { __bf16 b[2]; unsigned u; } p0, p1;
    p0.b[0] = (__bf16)v.x; p0.b[1] = (__bf16)v.y;
    p1.b[0] = (__bf16)v.z; p1.b[1] = (__bf16)v.w;
    unsigned short* base = sums + seg * C_DIM + lane * 4;
    asm volatile("global_atomic_pk_add_bf16 %0, %1, off"
                 :: "v"((unsigned long long)base), "v"(p0.u) : "memory");
    asm volatile("global_atomic_pk_add_bf16 %0, %1, off"
                 :: "v"((unsigned long long)(base + 2)), "v"(p1.u) : "memory");
}

// ---- inner product step: 8 N-tiles of WMMA against one A fragment ----
__device__ __forceinline__ void mma_row(const v16bf& afrag, v8f* acc,
                                        const unsigned short* __restrict__ wt,
                                        int kt, int lane) {
    int h  = lane >> 4;
    int nl = lane & 15;
    #pragma unroll
    for (int nt = 0; nt < 8; ++nt) {
        // B fragment (32x16 bf16): lane = N, element j = K = kt*32 + 16h + j
        const v16bf bfrag =
            *(const v16bf*)(wt + (size_t)(nt * 16 + nl) * K_DIM + kt * 32 + h * 16);
        acc[nt] = __builtin_amdgcn_wmma_f32_16x16x32_bf16(
            false, afrag, false, bfrag, (short)0, acc[nt], false, false);
    }
}

// ---- fused mean-finalize + GEMM: block = 8 waves = 128 rows x 128 cols ----
__global__ __launch_bounds__(256) void gemm_kernel(const float* __restrict__ x,
                                                   const unsigned short* __restrict__ sums,
                                                   const unsigned* __restrict__ cnt,
                                                   const unsigned short* __restrict__ wt,
                                                   float* __restrict__ out, int N) {
    int wave = threadIdx.x >> 5, lane = threadIdx.x & 31;
    int m0 = blockIdx.x * 128 + wave * 16;
    if (m0 >= N) return;                         // whole-wave uniform skip
    int mrow = m0 + (lane & 15);
    int h = lane >> 4;

    v8f acc[8];
    #pragma unroll
    for (int i = 0; i < 8; ++i) acc[i] = (v8f)0.0f;

    // edge-type slots 0..5: bf16 sums scaled by 1/max(cnt,1) in-register
    for (int t = 0; t < 6; ++t) {
        size_t seg = (size_t)mrow * 7 + (size_t)t;
        unsigned cv = cnt[seg];
        float scale = 1.0f / (float)(cv ? cv : 1u);
        const unsigned short* srow = sums + seg * C_DIM;
        #pragma unroll
        for (int kk = 0; kk < 4; ++kk) {
            // A fragment (16x32 bf16): element j = K = kc + 16h + j
            const v8bf lo = *(const v8bf*)(srow + kk * 32 + h * 16);
            const v8bf hi = *(const v8bf*)(srow + kk * 32 + h * 16 + 8);
            v16bf afrag;
            #pragma unroll
            for (int j = 0; j < 8; ++j) {
                afrag[j]     = (__bf16)((float)lo[j] * scale);
                afrag[8 + j] = (__bf16)((float)hi[j] * scale);
            }
            mma_row(afrag, acc, wt, t * 4 + kk, lane);
        }
    }
    // slot 6: self features straight from x (f32 -> bf16)
    {
        const float* xrow = x + (size_t)mrow * C_DIM;
        #pragma unroll
        for (int kk = 0; kk < 4; ++kk) {
            const float* pa = xrow + kk * 32 + h * 16;
            const float* pb = xrow + kk * 32 + h * 16 + 8;
            v16bf afrag;
            #pragma unroll
            for (int j = 0; j < 8; ++j) {
                afrag[j]     = (__bf16)pa[j];
                afrag[8 + j] = (__bf16)pb[j];
            }
            mma_row(afrag, acc, wt, 24 + kk, lane);
        }
    }

    // store C/D: VGPR v -> M = v + 8h, lanes 0-15 -> N = lane (per N-tile)
    int mb = m0 + h * 8;
    #pragma unroll
    for (int nt = 0; nt < 8; ++nt) {
        int n = nt * 16 + (lane & 15);
        #pragma unroll
        for (int v = 0; v < 8; ++v)
            out[(size_t)(mb + v) * C_DIM + n] = acc[nt][v];
    }
}

// ---- BN pass 1: per-channel sum / sumsq ----
__global__ __launch_bounds__(256) void bn_stats_kernel(const float* __restrict__ out,
                                                       float* __restrict__ stats, int N) {
    int c  = threadIdx.x & 127;
    int r0 = blockIdx.x * 2 + (threadIdx.x >> 7);
    float s = 0.0f, s2 = 0.0f;
    for (int r = r0; r < N; r += gridDim.x * 2) {
        float v = out[(size_t)r * C_DIM + c];
        s += v; s2 += v * v;
    }
    atomicAdd(&stats[c], s);
    atomicAdd(&stats[C_DIM + c], s2);
}

// ---- BN pass 2: normalize in place ----
__global__ __launch_bounds__(256) void bn_apply_kernel(float* __restrict__ out,
                                                       const float* __restrict__ stats,
                                                       const float* __restrict__ gamma,
                                                       const float* __restrict__ beta, int N) {
    size_t total = (size_t)N * C_DIM;
    float inv_n = 1.0f / (float)N;
    for (size_t i = (size_t)blockIdx.x * 256 + threadIdx.x; i < total;
         i += (size_t)gridDim.x * 256) {
        int c = (int)(i & 127);
        float mean = stats[c] * inv_n;
        float var  = stats[C_DIM + c] * inv_n - mean * mean;
        float v = out[i];
        out[i] = (v - mean) * rsqrtf(var + 1e-5f) * gamma[c] + beta[c];
    }
}

extern "C" void kernel_launch(void* const* d_in, const int* in_sizes, int n_in,
                              void* d_out, int out_size, void* d_ws, size_t ws_size,
                              hipStream_t stream) {
    const float* x     = (const float*)d_in[0];
    const int*   row   = (const int*)d_in[1];
    const int*   col   = (const int*)d_in[2];
    const int*   et    = (const int*)d_in[3];
    const float* W     = (const float*)d_in[4];
    const float* gamma = (const float*)d_in[5];
    const float* beta  = (const float*)d_in[6];
    int N = in_sizes[0] / C_DIM;
    int E = in_sizes[1];

    char* ws = (char*)d_ws;
    size_t szSums  = (size_t)N * 7 * C_DIM * 2;   // bf16
    size_t szCnt   = (size_t)N * 7 * 4;           // u32
    size_t szStats = 256 * 4;                     // f32 sum+sumsq
    unsigned short* sums  = (unsigned short*)ws;
    unsigned*       cnt   = (unsigned*)(ws + szSums);
    float*          stats = (float*)(ws + szSums + szCnt);
    unsigned short* wt    = (unsigned short*)(ws + szSums + szCnt + szStats);

    hipMemsetAsync(ws, 0, szSums + szCnt + szStats, stream);  // graph-capturable

    pack_w_kernel<<<(K_DIM * C_DIM + 255) / 256, 256, 0, stream>>>(W, wt);
    scatter_kernel<<<(E + 7) / 8, 256, 0, stream>>>(x, row, col, et, sums, cnt, E);
    gemm_kernel<<<(N + 127) / 128, 256, 0, stream>>>(x, sums, cnt, wt, (float*)d_out, N);
    bn_stats_kernel<<<1024, 256, 0, stream>>>((const float*)d_out, stats, N);
    bn_apply_kernel<<<2048, 256, 0, stream>>>((float*)d_out, stats, gamma, beta, N);
}
